// KoLeoLoss_29111288333061
// MI455X (gfx1250) — matrix operand order repaired and checked by reading, not compiled, not run on hardware
//
#include <hip/hip_runtime.h>
#include <hip/hip_bf16.h>

#define NROWS 16384
#define DIM   1024
#define EPSV  1e-8f

typedef __bf16 bf16_t;
typedef bf16_t v16bf __attribute__((ext_vector_type(16)));
typedef bf16_t v8bf  __attribute__((ext_vector_type(8)));
typedef float  v8f   __attribute__((ext_vector_type(8)));

// ---------------------------------------------------------------------------
// WMMA wrapper: D = A(16x32 bf16) * B(32x16 bf16) + C(16x16 f32)
// ---------------------------------------------------------------------------
__device__ __forceinline__ v8f wmma_bf16(v16bf a, v16bf b, v8f c) {
    return __builtin_amdgcn_wmma_f32_16x16x32_bf16(
        /*neg_a=*/false, a, /*neg_b=*/false, b,
        /*c_mod=*/(short)0, c, /*reuse_a=*/false, /*reuse_b=*/false);
}

// A fragment (16x32, row-major source, stride DIM):
//   lanes 0-15: M = lane, lanes 16-31: M = lane-16 (same rows, K-halves)
//   v[0..7]  = K = k + sub*8  + {0..7}
//   v[8..15] = K = k + 16 + sub*8 + {0..7}
__device__ __forceinline__ v16bf load_a_frag(const bf16_t* __restrict__ rowp,
                                             int k, int sub) {
    const v8bf* p = (const v8bf*)(rowp + k + sub * 8);   // 16B aligned
    v8bf a = p[0];        // K = k+sub*8 .. +7
    v8bf b = p[2];        // +16 elements
    return __builtin_shufflevector(a, b, 0, 1, 2, 3, 4, 5, 6, 7,
                                         8, 9, 10, 11, 12, 13, 14, 15);
}

// B fragment (32x16, source row = column of B since B = X^T tile):
//   lanes 0-15: N = lane, lanes 16-31: N = lane-16 (same K order as A fragment)
//   v[0..7] = K = k + sub*8 + {0..7}, v[8..15] = K = k + 16 + sub*8 + {0..7}
__device__ __forceinline__ v16bf load_b_frag(const bf16_t* __restrict__ colrowp,
                                             int k, int sub) {
    const v8bf* p = (const v8bf*)(colrowp + k + sub * 8);
    v8bf a = p[0];
    v8bf b = p[2];
    return __builtin_shufflevector(a, b, 0, 1, 2, 3, 4, 5, 6, 7,
                                         8, 9, 10, 11, 12, 13, 14, 15);
}

// ---------------------------------------------------------------------------
// Kernel 1: L2-normalize rows, emit rinv + bf16 hi/lo split of xn
// ---------------------------------------------------------------------------
__global__ __launch_bounds__(256)
void koleo_normalize(const float* __restrict__ x,
                     bf16_t* __restrict__ xh, bf16_t* __restrict__ xl,
                     float* __restrict__ rinv) {
    const int row = blockIdx.x;
    const float* xr = x + (size_t)row * DIM;

    float s = 0.0f;
    for (int k = threadIdx.x; k < DIM; k += 256) {
        float v = xr[k];
        s += v * v;
    }
    #pragma unroll
    for (int off = 16; off > 0; off >>= 1) s += __shfl_xor(s, off, 32);

    __shared__ float red[8];
    __shared__ float rinv_s;
    if ((threadIdx.x & 31) == 0) red[threadIdx.x >> 5] = s;
    __syncthreads();
    if (threadIdx.x == 0) {
        float tot = 0.0f;
        #pragma unroll
        for (int i = 0; i < 8; ++i) tot += red[i];
        float nrm = sqrtf(tot);
        float r = 1.0f / fmaxf(nrm, EPSV);
        rinv_s = r;
        rinv[row] = r;
    }
    __syncthreads();
    const float r = rinv_s;

    for (int k = threadIdx.x; k < DIM; k += 256) {
        float v = xr[k] * r;
        bf16_t h = (bf16_t)v;             // RNE
        float hv = (float)h;
        bf16_t l = (bf16_t)(v - hv);
        xh[(size_t)row * DIM + k] = h;
        xl[(size_t)row * DIM + k] = l;
    }
}

// ---------------------------------------------------------------------------
// Kernel 2: WMMA pairwise-similarity argmax (diagonal masked).
// Register-blocked: each wave owns TWO 16-row strips (M=32) and sweeps all
// columns 8 col-tiles (N=128) at a time; split-bf16 -> 6 WMMAs per B tile,
// B fragments amortized over both strips (13.3 B/lane/WMMA from L2).
// ---------------------------------------------------------------------------
__global__ __launch_bounds__(256)
void koleo_argmax(const bf16_t* __restrict__ xh, const bf16_t* __restrict__ xl,
                  int* __restrict__ nnidx) {
    const int lane = threadIdx.x & 31;
    const int wave = threadIdx.x >> 5;
    const int rowBase = (blockIdx.x * 8 + wave) * 32;   // 2 strips of 16
    const int sub  = lane >> 4;     // 0 / 1 : lane half
    const int nidx = lane & 15;

    float rmax[2][8];
    int   ridx[2][8];
    #pragma unroll
    for (int s = 0; s < 2; ++s)
        #pragma unroll
        for (int r = 0; r < 8; ++r) { rmax[s][r] = -2.0f; ridx[s][r] = 0; }

    // A lane rows for both strips
    const bf16_t* aph0 = xh + (size_t)(rowBase + nidx) * DIM;
    const bf16_t* apl0 = xl + (size_t)(rowBase + nidx) * DIM;
    const bf16_t* aph1 = xh + (size_t)(rowBase + 16 + nidx) * DIM;
    const bf16_t* apl1 = xl + (size_t)(rowBase + 16 + nidx) * DIM;

    const v8f vzero = {0.f, 0.f, 0.f, 0.f, 0.f, 0.f, 0.f, 0.f};

    for (int cg = 0; cg < NROWS; cg += 128) {      // 8 col-tiles per group
        v8f acc0[8], acc1[8];
        #pragma unroll
        for (int t = 0; t < 8; ++t) { acc0[t] = vzero; acc1[t] = vzero; }

        for (int k = 0; k < DIM; k += 32) {
            v16bf ah0 = load_a_frag(aph0, k, sub);
            v16bf al0 = load_a_frag(apl0, k, sub);
            v16bf ah1 = load_a_frag(aph1, k, sub);
            v16bf al1 = load_a_frag(apl1, k, sub);
            #pragma unroll
            for (int t = 0; t < 8; ++t) {
                const size_t brow = (size_t)(cg + t * 16 + nidx) * DIM;
                v16bf bh = load_b_frag(xh + brow, k, sub);
                v16bf bl = load_b_frag(xl + brow, k, sub);
                acc0[t] = wmma_bf16(ah0, bh, acc0[t]);
                acc1[t] = wmma_bf16(ah1, bh, acc1[t]);
                acc0[t] = wmma_bf16(ah0, bl, acc0[t]);
                acc1[t] = wmma_bf16(ah1, bl, acc1[t]);
                acc0[t] = wmma_bf16(al0, bh, acc0[t]);
                acc1[t] = wmma_bf16(al1, bh, acc1[t]);
            }
        }

        // running max/argmax update; D layout: (M = r + 8*sub, N = nidx)
        #pragma unroll
        for (int t = 0; t < 8; ++t) {
            const int col = cg + t * 16 + nidx;
            #pragma unroll
            for (int r = 0; r < 8; ++r) {
                const int row0 = rowBase + r + 8 * sub;
                const float v0 = acc0[t][r];
                if (col != row0 &&
                    (v0 > rmax[0][r] || (v0 == rmax[0][r] && col < ridx[0][r]))) {
                    rmax[0][r] = v0;
                    ridx[0][r] = col;
                }
                const int row1 = rowBase + 16 + r + 8 * sub;
                const float v1 = acc1[t][r];
                if (col != row1 &&
                    (v1 > rmax[1][r] || (v1 == rmax[1][r] && col < ridx[1][r]))) {
                    rmax[1][r] = v1;
                    ridx[1][r] = col;
                }
            }
        }
    }

    // reduce (val,idx) across the 16 lanes that share each row set
    #pragma unroll
    for (int s = 0; s < 2; ++s) {
        #pragma unroll
        for (int r = 0; r < 8; ++r) {
            float v = rmax[s][r];
            int   i = ridx[s][r];
            #pragma unroll
            for (int off = 1; off < 16; off <<= 1) {
                float ov = __shfl_xor(v, off, 16);
                int   oi = __shfl_xor(i, off, 16);
                if (ov > v || (ov == v && oi < i)) { v = ov; i = oi; }
            }
            if (nidx == 0) nnidx[rowBase + s * 16 + r + 8 * sub] = i;
        }
    }
}

// ---------------------------------------------------------------------------
// Kernel 3: exact f32 distance to NN, log, per-row (deterministic)
// ---------------------------------------------------------------------------
__global__ __launch_bounds__(256)
void koleo_dist(const float* __restrict__ x, const float* __restrict__ rinv,
                const int* __restrict__ nnidx, float* __restrict__ logd) {
    const int row = blockIdx.x;
    const int j   = nnidx[row];
    const float ri = rinv[row];
    const float rj = rinv[j];
    const float* xi = x + (size_t)row * DIM;
    const float* xj = x + (size_t)j * DIM;

    float s = 0.0f;
    for (int k = threadIdx.x; k < DIM; k += 256) {
        float d = xi[k] * ri - xj[k] * rj + EPSV;  // torch PairwiseDistance eps
        s += d * d;
    }
    #pragma unroll
    for (int off = 16; off > 0; off >>= 1) s += __shfl_xor(s, off, 32);

    __shared__ float red[8];
    if ((threadIdx.x & 31) == 0) red[threadIdx.x >> 5] = s;
    __syncthreads();
    if (threadIdx.x == 0) {
        float tot = 0.0f;
        #pragma unroll
        for (int i = 0; i < 8; ++i) tot += red[i];
        logd[row] = logf(sqrtf(tot) + EPSV);
    }
}

// ---------------------------------------------------------------------------
// Kernel 4: deterministic mean reduction -> loss = -mean(logd)
// ---------------------------------------------------------------------------
__global__ __launch_bounds__(256)
void koleo_final(const float* __restrict__ logd, float* __restrict__ out) {
    float s = 0.0f;
    for (int i = threadIdx.x; i < NROWS; i += 256) s += logd[i];
    #pragma unroll
    for (int off = 16; off > 0; off >>= 1) s += __shfl_xor(s, off, 32);

    __shared__ float red[8];
    if ((threadIdx.x & 31) == 0) red[threadIdx.x >> 5] = s;
    __syncthreads();
    if (threadIdx.x == 0) {
        float tot = 0.0f;
        #pragma unroll
        for (int i = 0; i < 8; ++i) tot += red[i];
        out[0] = -tot / (float)NROWS;
    }
}

// ---------------------------------------------------------------------------
extern "C" void kernel_launch(void* const* d_in, const int* in_sizes, int n_in,
                              void* d_out, int out_size, void* d_ws, size_t ws_size,
                              hipStream_t stream) {
    (void)in_sizes; (void)n_in; (void)out_size; (void)ws_size;

    const float* x = (const float*)d_in[0];
    float* out = (float*)d_out;

    // workspace layout
    char* ws = (char*)d_ws;
    const size_t halfBytes = (size_t)NROWS * DIM * sizeof(bf16_t);  // 32 MB
    bf16_t* xh   = (bf16_t*)ws;
    bf16_t* xl   = (bf16_t*)(ws + halfBytes);
    float*  rinv = (float*)(ws + 2 * halfBytes);
    int*    nni  = (int*)(ws + 2 * halfBytes + NROWS * sizeof(float));
    float*  logd = (float*)(ws + 2 * halfBytes + 2 * NROWS * sizeof(float));

    koleo_normalize<<<NROWS, 256, 0, stream>>>(x, xh, xl, rinv);
    koleo_argmax<<<NROWS / (32 * 8), 256, 0, stream>>>(xh, xl, nni);
    koleo_dist<<<NROWS, 256, 0, stream>>>(x, rinv, nni, logd);
    koleo_final<<<1, 256, 0, stream>>>(logd, out);
}
